// PrunedQwen3MoeSparseMoeBlock_15814069583889
// MI455X (gfx1250) — matrix operand order repaired and checked by reading, not compiled, not run on hardware
//
#include <hip/hip_runtime.h>
#include <hip/hip_bf16.h>

// ---------------------------------------------------------------------------
// PrunedQwen3MoeSparseMoeBlock for MI455X (gfx1250, wave32, WMMA).
//
// T=1024 tokens, H=2048, E=32 experts, I=384 (pruned), 2I=768, top_k=4.
//
//   k1: router (logits -> softmax -> top4 -> renorm) + per-expert token lists
//   k2: per (expert, 32-token tile) block: bf16 WMMA gate/up GEMM ->
//       SiLU*up -> bf16 WMMA down GEMM -> weighted atomic scatter into out.
//
// M-tile = 32 (two 16-row WMMA sub-tiles) so every B fragment load feeds two
// v_wmma_f32_16x16x32_bf16 issues: halves weight-side L2 traffic and halves
// the WMMA->VALU hazard overhead per MAC versus an M=16 tiling.
// ---------------------------------------------------------------------------

#define T_TOK 1024
#define HID   2048
#define NE    32
#define I2D   768
#define IPR   384
#define TOPK  4
#define MT    32          // tokens per expert tile (2 WMMA M sub-tiles)

typedef __attribute__((ext_vector_type(16))) __bf16 v16bf;
typedef __attribute__((ext_vector_type(8)))  float  v8f;

// ------------------------------- router ------------------------------------
// One wave32 per token; lane == expert index (E == 32 exactly).
__global__ __launch_bounds__(256) void moe_router_kernel(
    const float* __restrict__ x,       // [T, H]
    const float* __restrict__ gate_w,  // [E, H]
    int*   __restrict__ counts,        // [E]
    int*   __restrict__ tokIds,        // [E, T]
    float* __restrict__ tokW)          // [E, T]
{
    const int wave = (blockIdx.x * blockDim.x + threadIdx.x) >> 5;  // token id
    const int lane = threadIdx.x & 31;                              // expert id
    if (wave >= T_TOK) return;

    const float* xr = x + (size_t)wave * HID;
    const float* gr = gate_w + (size_t)lane * HID;

    // logit[lane] = dot(x[token], gate_w[lane]); broadcast x via shuffles.
    float acc = 0.0f;
    for (int h0 = 0; h0 < HID; h0 += 32) {
        float xv = xr[h0 + lane];
#pragma unroll
        for (int i = 0; i < 32; ++i) {
            float xi = __shfl(xv, i, 32);
            acc = fmaf(xi, gr[h0 + i], acc);
        }
    }

    // Wave-wide softmax (fp32).
    float m = acc;
#pragma unroll
    for (int off = 16; off > 0; off >>= 1) m = fmaxf(m, __shfl_xor(m, off, 32));
    float p = __expf(acc - m);
    float s = p;
#pragma unroll
    for (int off = 16; off > 0; off >>= 1) s += __shfl_xor(s, off, 32);
    const float prob = p / s;

    // Iterative top-4 argmax (smallest lane wins ties).
    float pw = prob;
    int   selLane[TOPK];
    float selP[TOPK];
    float wsum = 0.0f;
#pragma unroll
    for (int k = 0; k < TOPK; ++k) {
        float v  = pw;
        int   id = lane;
#pragma unroll
        for (int off = 16; off > 0; off >>= 1) {
            float ov  = __shfl_xor(v, off, 32);
            int   oid = __shfl_xor(id, off, 32);
            if (ov > v || (ov == v && oid < id)) { v = ov; id = oid; }
        }
        selLane[k] = id;
        selP[k]    = v;
        wsum      += v;
        if (lane == id) pw = -1.0f;   // remove winner
    }

    // Winning lanes append (token, renormalized weight) to their expert list.
#pragma unroll
    for (int k = 0; k < TOPK; ++k) {
        if (lane == selLane[k]) {
            float w = selP[k] / wsum;
            int pos = atomicAdd(&counts[lane], 1);
            tokIds[lane * T_TOK + pos] = wave;
            tokW[lane * T_TOK + pos]   = w;
        }
    }
}

// ---------------------------- expert MLP tile -------------------------------
// Block: 256 threads (8 waves). blockIdx.x = expert, blockIdx.y = 32-token tile.
__global__ __launch_bounds__(256) void moe_expert_kernel(
    const float* __restrict__ x,        // [T, H]
    const float* __restrict__ gup_w,    // [E, 2I, H]
    const float* __restrict__ down_w,   // [E, H, I]
    const int*   __restrict__ counts,   // [E]
    const int*   __restrict__ tokIds,   // [E, T]
    const float* __restrict__ tokW,     // [E, T]
    float* __restrict__ out)            // [T, H]
{
    const int e    = blockIdx.x;
    const int cnt  = counts[e];
    const int base = blockIdx.y * MT;
    if (base >= cnt) return;

    __shared__ int    sTok[MT];
    __shared__ float  sW[MT];
    __shared__ __bf16 sAct[MT * IPR];   // 24 KB bf16 activations

    const int tid = threadIdx.x;
    if (tid < MT) {
        int r = base + tid;
        if (r < cnt) {
            sTok[tid] = tokIds[e * T_TOK + r];
            sW[tid]   = tokW[e * T_TOK + r];
        } else {                        // pad with a valid token, zero weight
            sTok[tid] = tokIds[e * T_TOK + base];
            sW[tid]   = 0.0f;
        }
    }
    __syncthreads();

    const int wave = tid >> 5;
    const int lane = tid & 31;
    const int half = lane >> 4;         // K-half per 16-bit WMMA A/B layout
    const int lm   = lane & 15;

    // Two A-row blocks: sub-tile 0 -> tokens [0,16), sub-tile 1 -> [16,32).
    const float* xrow0 = x + (size_t)sTok[lm]      * HID;
    const float* xrow1 = x + (size_t)sTok[16 + lm] * HID;
    const float* Wge   = gup_w + (size_t)e * I2D * HID;

    // ---- gate/up GEMM: M=32, K=2048; wave owns 3 gate + 3 matching up
    // N-tiles, each with 2 M sub-tiles (12 accumulators).
    v8f acc[12];
#pragma unroll
    for (int j = 0; j < 12; ++j) acc[j] = (v8f){0,0,0,0,0,0,0,0};

    for (int k0 = 0; k0 < HID; k0 += 32) {
        // A fragments (16-bit A 16x32 layout): elems 0-7 hold K=half*8+0..7,
        // elems 8-15 hold K=16+half*8+0..7.
        v16bf a0, a1;
        {
            const float* p0 = xrow0 + k0 + half * 8;
            const float* p1 = xrow0 + k0 + 16 + half * 8;
            const float* q0 = xrow1 + k0 + half * 8;
            const float* q1 = xrow1 + k0 + 16 + half * 8;
#pragma unroll
            for (int i = 0; i < 8; ++i) { a0[i] = (__bf16)p0[i]; a0[8+i] = (__bf16)p1[i]; }
#pragma unroll
            for (int i = 0; i < 8; ++i) { a1[i] = (__bf16)q0[i]; a1[8+i] = (__bf16)q1[i]; }
        }

#pragma unroll
        for (int j = 0; j < 6; ++j) {
            const int n0 = (j < 3) ? (wave * 48 + 16 * j)
                                   : (IPR + wave * 48 + 16 * (j - 3));
            // B fragment (32x16): lane row = n0+lm, K = half*8+0..7, 16+half*8+0..7;
            // loaded once, consumed by BOTH M sub-tiles.
            const float* bp = Wge + (size_t)(n0 + lm) * HID + k0 + half * 8;
            v16bf b;
#pragma unroll
            for (int i = 0; i < 16; ++i) b[i] = (__bf16)bp[i + (i >= 8 ? 8 : 0)];
            acc[2*j+0] = __builtin_amdgcn_wmma_f32_16x16x32_bf16(
                false, a0, false, b, (short)0, acc[2*j+0], false, false);
            acc[2*j+1] = __builtin_amdgcn_wmma_f32_16x16x32_bf16(
                false, a1, false, b, (short)0, acc[2*j+1], false, false);
        }
    }

    // ---- SiLU(gate) * up -> bf16 LDS (gate tile j pairs with up tile j+3;
    // identical C layouts, purely element-wise in registers).
#pragma unroll
    for (int j = 0; j < 3; ++j) {
        const int n0 = wave * 48 + 16 * j;
#pragma unroll
        for (int s = 0; s < 2; ++s) {
#pragma unroll
            for (int r = 0; r < 8; ++r) {
                float g = acc[2*j + s][r];
                float u = acc[2*(j+3) + s][r];
                float av = (g / (1.0f + __expf(-g))) * u;
                const int mrow = s * 16 + r + 8 * half;   // C: M = r + 8*half
                sAct[mrow * IPR + n0 + lm] = (__bf16)av;
            }
        }
    }
    __syncthreads();

    // ---- down GEMM: M=32, K=384, N=2048; wave owns 256 output columns.
    for (int j = 0; j < 16; ++j) {
        const int n0 = wave * 256 + j * 16;
        v8f c0 = (v8f){0,0,0,0,0,0,0,0};
        v8f c1 = (v8f){0,0,0,0,0,0,0,0};
        for (int k0 = 0; k0 < IPR; k0 += 32) {
            v16bf a0, a1;
            {
                const __bf16* p0 = &sAct[lm        * IPR + k0 + half * 8];
                const __bf16* p1 = &sAct[lm        * IPR + k0 + 16 + half * 8];
                const __bf16* q0 = &sAct[(16 + lm) * IPR + k0 + half * 8];
                const __bf16* q1 = &sAct[(16 + lm) * IPR + k0 + 16 + half * 8];
#pragma unroll
                for (int i = 0; i < 8; ++i) { a0[i] = p0[i]; a0[8+i] = p1[i]; }
#pragma unroll
                for (int i = 0; i < 8; ++i) { a1[i] = q0[i]; a1[8+i] = q1[i]; }
            }
            // down_w[e, h, i]: row h = n0+lm, contraction over i; one B
            // fragment shared by both M sub-tiles.
            const float* bp = down_w + ((size_t)e * HID + n0 + lm) * IPR
                              + k0 + half * 8;
            v16bf b;
#pragma unroll
            for (int i = 0; i < 16; ++i) b[i] = (__bf16)bp[i + (i >= 8 ? 8 : 0)];
            c0 = __builtin_amdgcn_wmma_f32_16x16x32_bf16(
                false, a0, false, b, (short)0, c0, false, false);
            c1 = __builtin_amdgcn_wmma_f32_16x16x32_bf16(
                false, a1, false, b, (short)0, c1, false, false);
        }
        // Weighted scatter: out[tok, n] += rw * y (tokens appear in 4 experts).
#pragma unroll
        for (int r = 0; r < 8; ++r) {
            const int m0 = r + 8 * half;
            const int m1 = 16 + m0;
            atomicAdd(&out[(size_t)sTok[m0] * HID + n0 + lm], c0[r] * sW[m0]);
            atomicAdd(&out[(size_t)sTok[m1] * HID + n0 + lm], c1[r] * sW[m1]);
        }
    }
}

// ------------------------------- launcher -----------------------------------
extern "C" void kernel_launch(void* const* d_in, const int* in_sizes, int n_in,
                              void* d_out, int out_size, void* d_ws, size_t ws_size,
                              hipStream_t stream) {
    (void)in_sizes; (void)n_in; (void)out_size; (void)ws_size;
    const float* x      = (const float*)d_in[0];   // [1,1024,2048]
    const float* gate_w = (const float*)d_in[1];   // [32,2048]
    const float* gup_w  = (const float*)d_in[2];   // [32,768,2048]
    const float* down_w = (const float*)d_in[3];   // [32,2048,384]
    // d_in[4] = top_k scalar (fixed at 4)

    float* out = (float*)d_out;

    // Workspace layout: counts[E] | tokIds[E*T] | tokW[E*T]  (~256.1 KB)
    int*   counts = (int*)d_ws;
    int*   tokIds = counts + NE;
    float* tokW   = (float*)(tokIds + NE * T_TOK);

    hipMemsetAsync(out,    0, sizeof(float) * T_TOK * HID, stream);
    hipMemsetAsync(counts, 0, sizeof(int) * NE, stream);

    // Router: 1024 waves (one per token), 8 waves per block.
    moe_router_kernel<<<T_TOK / 8, 256, 0, stream>>>(x, gate_w, counts, tokIds, tokW);

    // Experts: 32 experts x up-to-32 32-token tiles (early-exit past count[e]).
    moe_expert_kernel<<<dim3(NE, T_TOK / MT), 256, 0, stream>>>(
        x, gup_w, down_w, counts, tokIds, tokW, out);
}
